// EdgeConvLayer_48713519071298
// MI455X (gfx1250) — compile-verified, matrix operand order checked by reading, not run on hardware
//
#include <hip/hip_runtime.h>
#include <hip/hip_bf16.h>

typedef __attribute__((ext_vector_type(16))) _Float16 v16h;
typedef __attribute__((ext_vector_type(8)))  _Float16 v8h;
typedef __attribute__((ext_vector_type(8)))  float    v8f;
typedef __attribute__((ext_vector_type(4)))  int      v4i;

#define BQ 256
#define PQ 128
#define FQ 7
#define HQ 256
#define KN 16
#define BIGD 1e18f
#define OUTC 264   // 256 + 7 + 1

#if __has_builtin(__builtin_amdgcn_global_load_async_to_lds_b128)
#define HAVE_ASYNC_LDS 1
#else
#define HAVE_ASYNC_LDS 0
#endif

typedef __attribute__((address_space(1))) v4i* gptr_v4i;  // global (AS1)
typedef __attribute__((address_space(3))) v4i* lptr_v4i;  // LDS (AS3)

// Copy 16 bytes (8 halfs) global -> LDS. Async DMA when available (ASYNCcnt),
// else a plain vectorized b128 load/store pair.
__device__ __forceinline__ void copy16_g2l(const _Float16* __restrict__ src,
                                           _Float16* __restrict__ dst) {
#if HAVE_ASYNC_LDS
    __builtin_amdgcn_global_load_async_to_lds_b128(
        (gptr_v4i)(src), (lptr_v4i)(dst), 0, 0);
#else
    *(v8h*)dst = *(const v8h*)src;
#endif
}
__device__ __forceinline__ void async_wait0() {
#if HAVE_ASYNC_LDS
#if __has_builtin(__builtin_amdgcn_s_wait_asynccnt)
    __builtin_amdgcn_s_wait_asynccnt(0);
#else
    asm volatile("s_wait_asynccnt 0x0" ::: "memory");
#endif
#endif
}

// A/B 16-bit fragments: per lane, element h (0..15) maps to two contiguous
// 8-half runs in K: [kb, kb+8) and [kb+16, kb+24) with kb = (lane>>4)*8.
__device__ __forceinline__ v16h frag_rowmajor(const _Float16* base) {
    union { v16h v; v8h h[2]; } u;
    u.h[0] = *(const v8h*)(base);
    u.h[1] = *(const v8h*)(base + 16);
    return u.v;
}
// Pre-packed layout: 16 contiguous halfs per (tile, lane).
__device__ __forceinline__ v16h frag_packed(const _Float16* base) {
    union { v16h v; v8h h[2]; } u;
    u.h[0] = *(const v8h*)(base);
    u.h[1] = *(const v8h*)(base + 8);
    return u.v;
}

// One-shot weight conversion fp32 -> f16, pre-packed in WMMA B-fragment
// lane layout: w1p[8192] = [nn][lane][16], w2p[65536] = [kk][nn][lane][16].
__global__ __launch_bounds__(256)
void pack_weights_kernel(const float* __restrict__ W1,
                         const float* __restrict__ W2,
                         _Float16* __restrict__ w1p,
                         _Float16* __restrict__ w2p) {
    int idx = blockIdx.x * 256 + threadIdx.x;
    if (idx < 16 * 32 * 16) {           // W1 pack (K padded 14 -> 32 with zeros)
        int h = idx & 15, lane = (idx >> 4) & 31, nn = idx >> 9;
        int n = nn * 16 + (lane & 15), kb = (lane >> 4) * 8;
        int k = (h < 8) ? (kb + h) : (16 + kb + (h - 8));
        w1p[idx] = (k < 2 * FQ) ? (_Float16)W1[k * HQ + n] : (_Float16)0.0f;
    }
    if (idx < 8 * 16 * 32 * 16) {       // W2 pack
        int h = idx & 15, lane = (idx >> 4) & 31;
        int nn = (idx >> 9) & 15, kk = idx >> 13;
        int n = nn * 16 + (lane & 15), kb = (lane >> 4) * 8;
        int k = (h < 8) ? (kb + h) : (16 + kb + (h - 8));
        w2p[idx] = (_Float16)W2[(kk * 32 + k) * HQ + n];
    }
}

template<bool USE_WS>
__global__ __launch_bounds__(128)
void edgeconv_fused_kernel(const float* __restrict__ events,
                           const float* __restrict__ W1,
                           const float* __restrict__ b1,
                           const float* __restrict__ W2,
                           const float* __restrict__ b2,
                           const _Float16* __restrict__ w1pack_g,
                           const _Float16* __restrict__ w2pack_g,
                           float* __restrict__ out) {
    __shared__ __align__(16) float sx[PQ], sy[PQ], sm[PQ];
    __shared__ __align__(16) float b1s[HQ], b2s[HQ];
    __shared__ __align__(16) _Float16 h1s[64 * 264];   // h1 tile, stride 264
    __shared__ __align__(16) _Float16 edA[64 * 40];    // edge tile, stride 40
    __shared__ __align__(16) _Float16 wbuf[16 * 32 * 16]; // W1pack, then W2 slices

    const int t  = threadIdx.x;
    const int w  = t >> 5;              // wave id 0..3 -> point within group
    const int L  = t & 31;              // lane in wave32
    const int b  = blockIdx.x >> 5;     // batch
    const int g  = blockIdx.x & 31;     // point group (4 points each)
    const int p  = g * 4 + w;           // this wave's point
    const long eb = (long)b * PQ * (FQ + 1);

    // ---- stage W1 fragments (async DMA from pre-packed ws when available) ----
    if (USE_WS) {
        #pragma unroll
        for (int i = 0; i < 8; ++i) {
            int c = (t + i * 128) * 8;
            copy16_g2l(w1pack_g + c, wbuf + c);
        }
    } else {
        for (int i = 0; i < 64; ++i) {
            int idx  = t + i * 128;
            int h    = idx & 15;
            int lane = (idx >> 4) & 31;
            int nn   = idx >> 9;
            int n    = nn * 16 + (lane & 15);
            int kb   = (lane >> 4) * 8;
            int k    = (h < 8) ? (kb + h) : (16 + kb + (h - 8));
            wbuf[idx] = (k < 2 * FQ) ? (_Float16)W1[k * HQ + n] : (_Float16)0.0f;
        }
    }
    // ---- stage coords/mask + biases ----
    {
        const float* ev = events + eb + (long)t * 8;
        sx[t] = ev[0]; sy[t] = ev[1]; sm[t] = ev[7];
        b1s[t] = b1[t]; b1s[t + 128] = b1[t + 128];
        b2s[t] = b2[t]; b2s[t + 128] = b2[t + 128];
    }
    if (USE_WS) async_wait0();
    __syncthreads();

    // ---- KNN for point p: one wave, registers only ----
    int myj = 0;   // lane (L&15)==k remembers neighbor index k
    {
        float px = sx[p], py = sy[p];
        float d[4];
        #pragma unroll
        for (int q = 0; q < 4; ++q) {
            int j = L + 32 * q;
            float dx = sx[j] - px, dy = sy[j] - py;
            float dd = dx * dx + dy * dy;             // sqrt is order-preserving
            if (sm[j] <= 0.5f || j == p) dd = BIGD;   // mask + self-exclusion
            d[q] = dd;
        }
        #pragma unroll
        for (int k = 0; k < KN; ++k) {
            float v = d[0]; int ix = L;
            #pragma unroll
            for (int q = 1; q < 4; ++q)
                if (d[q] < v) { v = d[q]; ix = L + 32 * q; }
            #pragma unroll
            for (int off = 16; off > 0; off >>= 1) {   // argmin, ties -> lower idx
                float ov = __shfl_xor(v, off, 32);
                int   oi = __shfl_xor(ix, off, 32);
                if (ov < v || (ov == v && oi < ix)) { v = ov; ix = oi; }
            }
            if ((L & 15) == k) myj = ix;
            if ((ix & 31) == L) {
                int q = ix >> 5;
                if (q == 0) d[0] = BIGD; else if (q == 1) d[1] = BIGD;
                else if (q == 2) d[2] = BIGD; else d[3] = BIGD;
            }
        }
    }

    // ---- build 64x32 edge tile (f16): [central(7) | neigh-central(7) | 0 pad] ----
    {
        int k   = L & 15;
        _Float16* er = edA + (w * 16 + k) * 40;
        const float* cp = events + eb + (long)p * 8;
        if (L < 16) {
            #pragma unroll
            for (int f = 0; f < 7; ++f) er[f] = (_Float16)cp[f];
        } else {
            const float* npf = events + eb + (long)myj * 8;
            #pragma unroll
            for (int f = 0; f < 7; ++f) er[7 + f] = (_Float16)(npf[f] - cp[f]);
            #pragma unroll
            for (int f = 14; f < 32; ++f) er[f] = (_Float16)0.0f;
        }
    }
    __syncthreads();

    // ---- GEMM1: edges(16x32) @ W1(32x256) per wave, bias+relu -> h1s ----
    {
        int m  = L & 15;
        int kb = (L >> 4) * 8;
        int n  = L & 15;
        v16h a = frag_rowmajor(edA + (w * 16 + m) * 40 + kb);
        #pragma unroll
        for (int nn = 0; nn < 16; ++nn) {
            v16h bf = frag_packed(wbuf + (nn * 32 + L) * 16);
            v8f c;
            float bias = b1s[nn * 16 + n];
            #pragma unroll
            for (int r = 0; r < 8; ++r) c[r] = bias;
            c = __builtin_amdgcn_wmma_f32_16x16x32_f16(false, a, false, bf,
                                                       (short)0, c, false, false);
            #pragma unroll
            for (int r = 0; r < 8; ++r) {
                int row = w * 16 + r + ((L >> 4) * 8);
                float vv = c[r];
                vv = vv > 0.0f ? vv : 0.0f;
                h1s[row * 264 + nn * 16 + n] = (_Float16)vv;
            }
        }
    }

    // ---- GEMM2: h1(16x256) @ W2(256x256) per wave, K staged 32 at a time ----
    v8f acc[16];
    {
        int n = L & 15;
        #pragma unroll
        for (int nn = 0; nn < 16; ++nn) {
            float bias = b2s[nn * 16 + n];
            #pragma unroll
            for (int r = 0; r < 8; ++r) acc[nn][r] = bias;
        }
    }
    for (int kk = 0; kk < 8; ++kk) {
        __syncthreads();                    // prior readers of wbuf are done
        if (USE_WS) {                        // async DMA of pre-packed k-slice
            #pragma unroll
            for (int i = 0; i < 8; ++i) {
                int c = (t + i * 128) * 8;
                copy16_g2l(w2pack_g + kk * 8192 + c, wbuf + c);
            }
            async_wait0();
        } else {                             // in-kernel scalar pack fallback
            for (int i = 0; i < 64; ++i) {
                int idx  = t + i * 128;
                int h    = idx & 15;
                int lane = (idx >> 4) & 31;
                int nn   = idx >> 9;
                int n    = nn * 16 + (lane & 15);
                int kb   = (lane >> 4) * 8;
                int k    = (h < 8) ? (kb + h) : (16 + kb + (h - 8));
                wbuf[idx] = (_Float16)W2[(kk * 32 + k) * HQ + n];
            }
        }
        __syncthreads();
        int m  = L & 15;
        int kb = (L >> 4) * 8;
        v16h a = frag_rowmajor(h1s + (w * 16 + m) * 264 + kk * 32 + kb);
        #pragma unroll
        for (int nn = 0; nn < 16; ++nn) {
            v16h bf = frag_packed(wbuf + (nn * 32 + L) * 16);
            acc[nn] = __builtin_amdgcn_wmma_f32_16x16x32_f16(false, a, false, bf,
                                                             (short)0, acc[nn],
                                                             false, false);
        }
    }

    // ---- relu, mean over K=16 rows, leaky-relu, mask, write ----
    {
        float mk = sm[p];
        const float* cp = events + eb + (long)p * 8;
        float* ob = out + ((long)b * PQ + p) * OUTC;
        #pragma unroll
        for (int nn = 0; nn < 16; ++nn) {
            float s = 0.0f;
            #pragma unroll
            for (int r = 0; r < 8; ++r) {
                float vv = acc[nn][r];
                s += (vv > 0.0f ? vv : 0.0f);
            }
            s += __shfl_xor(s, 16, 32);     // combine M=0..7 with M=8..15
            if (L < 16) {
                float aggv = s * (1.0f / 16.0f);
                float lv = aggv > 0.0f ? aggv : 0.2f * aggv;
                ob[nn * 16 + L] = lv * mk;
            }
        }
        if (L < 8) {
            if (L < 7) {
                float fv = cp[L];
                float lv = fv > 0.0f ? fv : 0.2f * fv;
                ob[256 + L] = lv * mk;
            } else {
                ob[263] = mk;
            }
        }
    }
}

extern "C" void kernel_launch(void* const* d_in, const int* in_sizes, int n_in,
                              void* d_out, int out_size, void* d_ws, size_t ws_size,
                              hipStream_t stream) {
    (void)in_sizes; (void)n_in; (void)out_size;
    const float* events = (const float*)d_in[0];
    const float* W1     = (const float*)d_in[1];
    const float* b1     = (const float*)d_in[2];
    const float* W2     = (const float*)d_in[3];
    const float* b2     = (const float*)d_in[4];
    float* out          = (float*)d_out;

    const size_t W1P_HALFS = 16 * 32 * 16;          // 8192
    const size_t W2P_HALFS = 8 * 16 * 32 * 16;      // 65536
    const size_t need = (W1P_HALFS + W2P_HALFS) * 2;

    if (d_ws != nullptr && ws_size >= need) {
        _Float16* w1p = (_Float16*)d_ws;
        _Float16* w2p = w1p + W1P_HALFS;
        pack_weights_kernel<<<dim3((unsigned)(W2P_HALFS / 256)), dim3(256), 0,
                              stream>>>(W1, W2, w1p, w2p);
        edgeconv_fused_kernel<true><<<dim3(BQ * (PQ / 4)), dim3(128), 0, stream>>>(
            events, W1, b1, W2, b2, w1p, w2p, out);
    } else {
        edgeconv_fused_kernel<false><<<dim3(BQ * (PQ / 4)), dim3(128), 0, stream>>>(
            events, W1, b1, W2, b2, nullptr, nullptr, out);
    }
}
